// PoolAttention_9878424781011
// MI455X (gfx1250) — matrix operand order repaired and checked by reading, not compiled, not run on hardware
//
#include <hip/hip_runtime.h>
#include <hip/hip_bf16.h>
#include <math.h>

// ---------------- problem constants ----------------
constexpr int Bn      = 8;
constexpr int NTOK    = 1025;
constexpr int CDIM    = 768;
constexpr int NHEADS  = 12;
constexpr int HDIM    = 64;
constexpr float ATT_SCALE = 0.125f;          // 64^-0.5
constexpr float MASK_FILL = -10000.0f;
constexpr float LN_EPS    = 1e-5f;
constexpr int MROWS   = Bn * NTOK;           // 8200

typedef __attribute__((ext_vector_type(16))) _Float16 v16h;
typedef __attribute__((ext_vector_type(8)))  float    v8f;

// ---------------- fragment-layout helpers (CDNA5 wave32, ISA 7.12.2) ----------
// A 16x32 f16 fragment: element (m,k) lives in lane = m + 16*((k>>3)&1),
// half slot h = ((k>>4)<<3) + (k&7).  For even k: (k,k+1) -> (h,h+1), same lane.
__device__ __forceinline__ int a_frag_idx(int m, int k) {
  return (m + (((k >> 3) & 1) << 4)) * 16 + (((k >> 4) << 3) + (k & 7));
}
// B 32x16 f16 fragment: element (k,n) lives in lane = n + 16*((k>>3)&1),
// half slot h = ((k>>4)<<3) + (k&7)  (same k order as A).
__device__ __forceinline__ int b_frag_idx(int k, int n) {
  return (n + (((k >> 3) & 1) << 4)) * 16 + (((k >> 4) << 3) + (k & 7));
}

__device__ __forceinline__ unsigned pack2(float a, float b) {
  union { _Float16 h[2]; unsigned u; } p;
  p.h[0] = (_Float16)a; p.h[1] = (_Float16)b;
  return p.u;
}

__device__ __forceinline__ v8f wmma32(v16h a, v16h b, v8f c) {
  return __builtin_amdgcn_wmma_f32_16x16x32_f16(false, a, false, b, (short)0, c,
                                                false, false);
}

// ---------------- kernel 1: QKV projection GEMM --------------------------------
constexpr int GT_M = 128, GT_N = 64, GT_K = 32;

__global__ __launch_bounds__(256)
void qkv_gemm_kernel(const float* __restrict__ x,
                     const float* __restrict__ Wq, const float* __restrict__ Wk,
                     const float* __restrict__ Wv,
                     float* __restrict__ qraw, float* __restrict__ kraw,
                     float* __restrict__ vraw) {
  __shared__ v16h AF[8 * 32];            // 8 strips x 32 lanes  (8 KB)
  __shared__ v16h BF[4 * 32];            // 4 n-tiles x 32 lanes (4 KB)
  unsigned* afu = (unsigned*)AF;
  unsigned* bfu = (unsigned*)BF;
  const float* W = (blockIdx.z == 0) ? Wq : (blockIdx.z == 1 ? Wk : Wv);
  float* outp    = (blockIdx.z == 0) ? qraw : (blockIdx.z == 1 ? kraw : vraw);
  const int tid  = threadIdx.x;
  const int wave = tid >> 5;
  const int lane = tid & 31;
  const int m0 = blockIdx.x * GT_M;
  const int n0 = blockIdx.y * GT_N;
  v8f acc[4] = {};
  for (int k0 = 0; k0 < CDIM; k0 += GT_K) {
    __syncthreads();
    for (int i = tid; i < GT_M * (GT_K / 2); i += 256) {   // A: 128 rows x 16 k-pairs
      const int r = i >> 4;
      const int k = (i & 15) << 1;
      const int gm = m0 + r;
      float v0 = 0.f, v1 = 0.f;
      if (gm < MROWS) {
        const float* p = &x[(size_t)gm * CDIM + k0 + k];
        v0 = p[0]; v1 = p[1];
      }
      afu[(((r >> 4) * 512) + a_frag_idx(r & 15, k)) >> 1] = pack2(v0, v1);
    }
    for (int i = tid; i < (GT_K / 2) * GT_N; i += 256) {   // B[k][n] = W[n0+n][k0+k]
      const int n = i >> 4;
      const int k = (i & 15) << 1;
      const float* p = &W[(size_t)(n0 + n) * CDIM + k0 + k];
      bfu[(((n >> 4) * 512) + b_frag_idx(k, n & 15)) >> 1] = pack2(p[0], p[1]);
    }
    __syncthreads();
    // hoist all operand loads so DS loads overlap; single wait before WMMA chain
    const v16h a  = AF[wave * 32 + lane];
    const v16h b0 = BF[0 * 32 + lane];
    const v16h b1 = BF[1 * 32 + lane];
    const v16h b2 = BF[2 * 32 + lane];
    const v16h b3 = BF[3 * 32 + lane];
    acc[0] = wmma32(a, b0, acc[0]);
    acc[1] = wmma32(a, b1, acc[1]);
    acc[2] = wmma32(a, b2, acc[2]);
    acc[3] = wmma32(a, b3, acc[3]);
  }
  const int nlo = lane & 15, hsl = lane >> 4;
  const int h = n0 / HDIM;                               // GT_N == HDIM
  size_t rowbase[8];
  bool   rowok[8];
#pragma unroll
  for (int r = 0; r < 8; ++r) {                          // hoist row mapping (8 divs)
    const int gm = m0 + wave * 16 + r + (hsl << 3);
    rowok[r] = (gm < MROWS);
    const int bb = gm / NTOK;
    const int nn = gm - bb * NTOK;
    rowbase[r] = (((size_t)bb * NHEADS + h) * NTOK + nn) * HDIM;
  }
#pragma unroll
  for (int t = 0; t < 4; ++t) {
#pragma unroll
    for (int r = 0; r < 8; ++r) {
      if (rowok[r]) outp[rowbase[r] + t * 16 + nlo] = acc[t][r];
    }
  }
}

// ---------------- kernel 2: depthwise 3x3 pool + LayerNorm ---------------------
__global__ __launch_bounds__(64)
void pool_ln_kernel(const float* __restrict__ qraw, const float* __restrict__ kraw,
                    const float* __restrict__ vraw,
                    const float* __restrict__ wqc, const float* __restrict__ wkc,
                    const float* __restrict__ wvc,
                    const float* __restrict__ gq, const float* __restrict__ bq,
                    const float* __restrict__ gk, const float* __restrict__ bk,
                    const float* __restrict__ gv, const float* __restrict__ bv,
                    float* __restrict__ qp, float* __restrict__ kp,
                    float* __restrict__ vp) {
  const int which = blockIdx.y;
  const float* raw = (which == 0) ? qraw : (which == 1 ? kraw : vraw);
  const float* w   = (which == 0) ? wqc  : (which == 1 ? wkc  : wvc);
  const float* g   = (which == 0) ? gq   : (which == 1 ? gk   : gv);
  const float* be  = (which == 0) ? bq   : (which == 1 ? bk   : bv);
  float* outp      = (which == 0) ? qp   : (which == 1 ? kp   : vp);

  const int d  = threadIdx.x;
  const int n  = blockIdx.x % NTOK;
  const int bh = blockIdx.x / NTOK;
  const float* base = raw + (size_t)bh * NTOK * HDIM;

  float val;
  if (n == 0) {
    val = base[d];                                      // cls token: no conv
  } else {
    const int s = n - 1;
    const int r = s >> 5, c = s & 31;
    float acc = 0.f;
#pragma unroll
    for (int dr = -1; dr <= 1; ++dr) {
#pragma unroll
      for (int dc = -1; dc <= 1; ++dc) {
        const int rr = r + dr, cc = c + dc;
        if (rr >= 0 && rr < 32 && cc >= 0 && cc < 32)
          acc += base[(size_t)(1 + rr * 32 + cc) * HDIM + d] *
                 w[d * 9 + (dr + 1) * 3 + (dc + 1)];
      }
    }
    val = acc;
  }
  // LayerNorm over the 64 channels
  __shared__ float red[64];
  __shared__ float stats[2];
  red[d] = val; __syncthreads();
  for (int off = 32; off > 0; off >>= 1) { if (d < off) red[d] += red[d + off]; __syncthreads(); }
  if (d == 0) stats[0] = red[0] * (1.f / 64.f);
  __syncthreads();
  const float mean = stats[0];
  const float dv = val - mean;
  red[d] = dv * dv; __syncthreads();
  for (int off = 32; off > 0; off >>= 1) { if (d < off) red[d] += red[d + off]; __syncthreads(); }
  if (d == 0) stats[1] = red[0] * (1.f / 64.f);
  __syncthreads();
  outp[((size_t)bh * NTOK + n) * HDIM + d] =
      dv * rsqrtf(stats[1] + LN_EPS) * g[d] + be[d];
}

// ---------------- kernel 3: scores = scale * Q K^T, masked ---------------------
__global__ __launch_bounds__(256)
void scores_kernel(const float* __restrict__ qp, const float* __restrict__ kp,
                   const int* __restrict__ mask, float* __restrict__ attn) {
  __shared__ v16h QF[4 * 2 * 32];   // [strip][kstep][lane]   (8 KB)
  __shared__ v16h KF[2 * 4 * 32];   // [kstep][ntile][lane]   (8 KB)
  unsigned* qfu = (unsigned*)QF;
  unsigned* kfu = (unsigned*)KF;
  const int tid  = threadIdx.x;
  const int wave = tid >> 5;
  const int lane = tid & 31;
  const int bh = blockIdx.z;
  const int q0 = blockIdx.x * 64;
  const int k0 = blockIdx.y * 64;
  const float* qb = qp + (size_t)bh * NTOK * HDIM;
  const float* kb = kp + (size_t)bh * NTOK * HDIM;
  for (int i = tid; i < 64 * 32; i += 256) {            // Q: 64 rows x 32 d-pairs
    const int qi = i >> 5;
    const int d  = (i & 31) << 1;
    const int gq = q0 + qi;
    float v0 = 0.f, v1 = 0.f;
    if (gq < NTOK) { const float* p = &qb[(size_t)gq * HDIM + d]; v0 = p[0]; v1 = p[1]; }
    qfu[((((qi >> 4) * 2 + (d >> 5)) * 512) + a_frag_idx(qi & 15, d & 31)) >> 1] =
        pack2(v0, v1);
  }
  for (int i = tid; i < 64 * 32; i += 256) {            // K^T: 64 cols x 32 d-pairs
    const int kj = i >> 5;
    const int d  = (i & 31) << 1;
    const int gk = k0 + kj;
    float v0 = 0.f, v1 = 0.f;
    if (gk < NTOK) { const float* p = &kb[(size_t)gk * HDIM + d]; v0 = p[0]; v1 = p[1]; }
    kfu[((((d >> 5) * 4 + (kj >> 4)) * 512) + b_frag_idx(d & 31, kj & 15)) >> 1] =
        pack2(v0, v1);
  }
  __syncthreads();
  const int tr  = wave >> 1;            // 4 tile-rows
  const int tcb = (wave & 1) * 2;       // 2 tile-cols each
  v8f acc[2] = {};
#pragma unroll
  for (int ks = 0; ks < 2; ++ks) {
    const v16h a  = QF[(tr * 2 + ks) * 32 + lane];
    const v16h b0 = KF[(ks * 4 + tcb + 0) * 32 + lane];
    const v16h b1 = KF[(ks * 4 + tcb + 1) * 32 + lane];
    acc[0] = wmma32(a, b0, acc[0]);
    acc[1] = wmma32(a, b1, acc[1]);
  }
  const int nlo = lane & 15, hsl = lane >> 4;
  const int bb = bh / NHEADS;
  float* arow = attn + (size_t)bh * NTOK * NTOK;
  int  rowq[8];
  bool mok[8];
#pragma unroll
  for (int r = 0; r < 8; ++r) {                          // hoist per-row mask load
    const int gq = q0 + tr * 16 + r + (hsl << 3);
    rowq[r] = gq;
    mok[r]  = (gq < NTOK) ? (mask[bb * NTOK + gq] != 0) : false;
  }
#pragma unroll
  for (int t = 0; t < 2; ++t) {
#pragma unroll
    for (int r = 0; r < 8; ++r) {
      const int gq = rowq[r];
      const int gk = k0 + (tcb + t) * 16 + nlo;
      if (gq < NTOK && gk < NTOK) {
        const float v = mok[r] ? (acc[t][r] * ATT_SCALE) : MASK_FILL;
        arow[(size_t)gq * NTOK + gk] = v;
      }
    }
  }
}

// ---------------- kernel 4: row softmax (in place) -----------------------------
__global__ __launch_bounds__(256)
void softmax_kernel(float* __restrict__ attn) {
  float* p = attn + (size_t)blockIdx.x * NTOK;
  const int tid = threadIdx.x;
  __shared__ float red[256];
  float mx = -3.4e38f;
  for (int i = tid; i < NTOK; i += 256) mx = fmaxf(mx, p[i]);
  red[tid] = mx; __syncthreads();
  for (int off = 128; off > 0; off >>= 1) { if (tid < off) red[tid] = fmaxf(red[tid], red[tid + off]); __syncthreads(); }
  mx = red[0]; __syncthreads();
  float sum = 0.f;
  for (int i = tid; i < NTOK; i += 256) { const float e = __expf(p[i] - mx); p[i] = e; sum += e; }
  red[tid] = sum; __syncthreads();
  for (int off = 128; off > 0; off >>= 1) { if (tid < off) red[tid] += red[tid + off]; __syncthreads(); }
  const float inv = 1.f / red[0];
  for (int i = tid; i < NTOK; i += 256) p[i] *= inv;
}

// ---------------- kernel 5: out = attn @ V (+ residual q) ----------------------
__global__ __launch_bounds__(256)
void av_kernel(const float* __restrict__ attn, const float* __restrict__ vp,
               const float* __restrict__ qpool, float* __restrict__ outpre) {
  __shared__ v16h AFa[4 * 32];       // [strip][lane]  64x32 tile
  __shared__ v16h BFa[4 * 32];       // [ntile][lane]  32x64 tile
  unsigned* afu = (unsigned*)AFa;
  unsigned* bfu = (unsigned*)BFa;
  const int tid  = threadIdx.x;
  const int wave = tid >> 5;
  const int lane = tid & 31;
  const int bh = blockIdx.z;
  const int q0 = blockIdx.x * 64;
  const float* ab = attn  + (size_t)bh * NTOK * NTOK;
  const float* vb = vp    + (size_t)bh * NTOK * HDIM;
  const float* qb = qpool + (size_t)bh * NTOK * HDIM;
  const int tr  = wave >> 1;
  const int tcb = (wave & 1) * 2;
  v8f acc[2] = {};
  for (int k0 = 0; k0 < 1056; k0 += 32) {               // 33 steps cover 1025 keys
    __syncthreads();
    for (int i = tid; i < 64 * 16; i += 256) {          // A: probs, 64 rows x 16 k-pairs
      const int qi = i >> 4;
      const int kk = (i & 15) << 1;
      const int gq = q0 + qi, gk = k0 + kk;
      float v0 = 0.f, v1 = 0.f;
      if (gq < NTOK) {
        if (gk     < NTOK) v0 = ab[(size_t)gq * NTOK + gk];
        if (gk + 1 < NTOK) v1 = ab[(size_t)gq * NTOK + gk + 1];
      }
      afu[(((qi >> 4) * 512) + a_frag_idx(qi & 15, kk)) >> 1] = pack2(v0, v1);
    }
    for (int i = tid; i < 16 * 64; i += 256) {          // B: V, 64 cols x 16 k-pairs
      const int n  = i >> 4;
      const int kk = (i & 15) << 1;
      const int gk = k0 + kk;
      float v0 = 0.f, v1 = 0.f;
      if (gk     < NTOK) v0 = vb[(size_t)gk * HDIM + n];
      if (gk + 1 < NTOK) v1 = vb[(size_t)(gk + 1) * HDIM + n];
      bfu[(((n >> 4) * 512) + b_frag_idx(kk, n & 15)) >> 1] = pack2(v0, v1);
    }
    __syncthreads();
    const v16h a  = AFa[tr * 32 + lane];
    const v16h b0 = BFa[(tcb + 0) * 32 + lane];
    const v16h b1 = BFa[(tcb + 1) * 32 + lane];
    acc[0] = wmma32(a, b0, acc[0]);
    acc[1] = wmma32(a, b1, acc[1]);
  }
  const int nlo = lane & 15, hsl = lane >> 4;
  const int bb = bh / NHEADS, hh = bh % NHEADS;
  int    rowq[8];
  size_t obase[8];
#pragma unroll
  for (int r = 0; r < 8; ++r) {                          // hoist row mapping
    const int gq = q0 + tr * 16 + r + (hsl << 3);
    rowq[r]  = gq;
    obase[r] = ((size_t)bb * NTOK + gq) * CDIM + hh * HDIM;
  }
#pragma unroll
  for (int t = 0; t < 2; ++t) {
#pragma unroll
    for (int r = 0; r < 8; ++r) {
      const int gq = rowq[r];
      const int d  = (tcb + t) * 16 + nlo;
      if (gq < NTOK) {
        float v = acc[t][r];
        if (gq >= 1) v += qb[(size_t)gq * HDIM + d];    // residual on non-cls
        outpre[obase[r] + d] = v;
      }
    }
  }
}

// ---------------- kernel 6: output projection + bias ---------------------------
__global__ __launch_bounds__(256)
void proj_gemm_kernel(const float* __restrict__ A, const float* __restrict__ Wp,
                      const float* __restrict__ bp, float* __restrict__ out) {
  __shared__ v16h AF[8 * 32];
  __shared__ v16h BF[4 * 32];
  unsigned* afu = (unsigned*)AF;
  unsigned* bfu = (unsigned*)BF;
  const int tid  = threadIdx.x;
  const int wave = tid >> 5;
  const int lane = tid & 31;
  const int m0 = blockIdx.x * GT_M;
  const int n0 = blockIdx.y * GT_N;
  v8f acc[4] = {};
  for (int k0 = 0; k0 < CDIM; k0 += GT_K) {
    __syncthreads();
    for (int i = tid; i < GT_M * (GT_K / 2); i += 256) {
      const int r = i >> 4;
      const int k = (i & 15) << 1;
      const int gm = m0 + r;
      float v0 = 0.f, v1 = 0.f;
      if (gm < MROWS) {
        const float* p = &A[(size_t)gm * CDIM + k0 + k];
        v0 = p[0]; v1 = p[1];
      }
      afu[(((r >> 4) * 512) + a_frag_idx(r & 15, k)) >> 1] = pack2(v0, v1);
    }
    for (int i = tid; i < (GT_K / 2) * GT_N; i += 256) {
      const int n = i >> 4;
      const int k = (i & 15) << 1;
      const float* p = &Wp[(size_t)(n0 + n) * CDIM + k0 + k];
      bfu[(((n >> 4) * 512) + b_frag_idx(k, n & 15)) >> 1] = pack2(p[0], p[1]);
    }
    __syncthreads();
    const v16h a  = AF[wave * 32 + lane];
    const v16h b0 = BF[0 * 32 + lane];
    const v16h b1 = BF[1 * 32 + lane];
    const v16h b2 = BF[2 * 32 + lane];
    const v16h b3 = BF[3 * 32 + lane];
    acc[0] = wmma32(a, b0, acc[0]);
    acc[1] = wmma32(a, b1, acc[1]);
    acc[2] = wmma32(a, b2, acc[2]);
    acc[3] = wmma32(a, b3, acc[3]);
  }
  const int nlo = lane & 15, hsl = lane >> 4;
  size_t rowbase[8];
  bool   rowok[8];
#pragma unroll
  for (int r = 0; r < 8; ++r) {
    const int gm = m0 + wave * 16 + r + (hsl << 3);
    rowok[r]   = (gm < MROWS);
    rowbase[r] = (size_t)gm * CDIM + n0;
  }
#pragma unroll
  for (int t = 0; t < 4; ++t) {
#pragma unroll
    for (int r = 0; r < 8; ++r) {
      const int cn = t * 16 + nlo;
      if (rowok[r]) out[rowbase[r] + cn] = acc[t][r] + bp[n0 + cn];
    }
  }
}

// ---------------- launcher -----------------------------------------------------
extern "C" void kernel_launch(void* const* d_in, const int* in_sizes, int n_in,
                              void* d_out, int out_size, void* d_ws, size_t ws_size,
                              hipStream_t stream) {
  (void)in_sizes; (void)n_in; (void)out_size; (void)ws_size;
  const float* x    = (const float*)d_in[0];
  const int*   mask = (const int*)d_in[1];
  const float* Wq   = (const float*)d_in[2];
  const float* Wk   = (const float*)d_in[3];
  const float* Wv   = (const float*)d_in[4];
  const float* pqw  = (const float*)d_in[5];
  const float* pkw  = (const float*)d_in[6];
  const float* pvw  = (const float*)d_in[7];
  const float* gq   = (const float*)d_in[8];
  const float* bq   = (const float*)d_in[9];
  const float* gk   = (const float*)d_in[10];
  const float* bk   = (const float*)d_in[11];
  const float* gv   = (const float*)d_in[12];
  const float* bv   = (const float*)d_in[13];
  const float* Wp   = (const float*)d_in[14];
  const float* bp   = (const float*)d_in[15];

  // workspace layout: 7 buffers of B*H*N*D floats (~176 MB total)
  const size_t SZ = (size_t)Bn * NHEADS * NTOK * HDIM;   // 6,297,600
  float* qraw   = (float*)d_ws;
  float* kraw   = qraw + SZ;
  float* vraw   = kraw + SZ;
  float* qpool  = vraw + SZ;
  float* kpool  = qpool + SZ;
  float* vpool  = kpool + SZ;
  float* outpre = vpool + SZ;

  float* out  = (float*)d_out;
  float* attn = out + (size_t)Bn * NTOK * CDIM;          // second return value

  dim3 g1((MROWS + GT_M - 1) / GT_M, CDIM / GT_N, 3);
  qkv_gemm_kernel<<<g1, 256, 0, stream>>>(x, Wq, Wk, Wv, qraw, kraw, vraw);

  dim3 g2(Bn * NHEADS * NTOK, 3);
  pool_ln_kernel<<<g2, 64, 0, stream>>>(qraw, kraw, vraw, pqw, pkw, pvw,
                                        gq, bq, gk, bk, gv, bv,
                                        qpool, kpool, vpool);

  dim3 g3((NTOK + 63) / 64, (NTOK + 63) / 64, Bn * NHEADS);
  scores_kernel<<<g3, 256, 0, stream>>>(qpool, kpool, mask, attn);

  softmax_kernel<<<dim3(Bn * NHEADS * NTOK), 256, 0, stream>>>(attn);

  dim3 g5((NTOK + 63) / 64, 1, Bn * NHEADS);
  av_kernel<<<g5, 256, 0, stream>>>(attn, vpool, qpool, outpre);

  dim3 g6((MROWS + GT_M - 1) / GT_M, CDIM / GT_N);
  proj_gemm_kernel<<<g6, 256, 0, stream>>>(outpre, Wp, bp, out);
}
